// KNRM_4518305595938
// MI455X (gfx1250) — matrix operand order repaired and checked by reading, not compiled, not run on hardware
//
#include <hip/hip_runtime.h>
#include <hip/hip_bf16.h>
#include <math.h>

typedef __attribute__((ext_vector_type(16))) _Float16 v16h;
typedef __attribute__((ext_vector_type(8)))  _Float16 v8h;
typedef __attribute__((ext_vector_type(8)))  float    v8f;

#define B_  256
#define Q_  32
#define D_  512
#define E_  300
#define EP_ 320      // E padded to multiple of 32 (WMMA K)
#define K_  11

// ---------------------------------------------------------------------------
// Kernel 1: gather embedding row, L2-normalize, store padded f16 row + mask.
// One wave (32 lanes) per row; 10 strided elements per lane covers 320.
// ---------------------------------------------------------------------------
__global__ __launch_bounds__(256)
void knrm_normalize_kernel(const int* __restrict__ tokens,
                           const float* __restrict__ emb,
                           _Float16* __restrict__ outn,
                           float* __restrict__ mask,
                           int nrows)
{
    const int lane = threadIdx.x & 31;
    const int wave = threadIdx.x >> 5;
    const int row  = blockIdx.x * 8 + wave;
    if (row >= nrows) return;

    const int tok = tokens[row];
    if (lane == 0) mask[row] = (tok > 0) ? 1.0f : 0.0f;

    const float* src = emb + (size_t)tok * E_;
    float v[10];
    float ss = 0.0f;
#pragma unroll
    for (int i = 0; i < 10; ++i) {
        const int e = lane + i * 32;
        const float x = (e < E_) ? src[e] : 0.0f;
        v[i] = x;
        ss += x * x;
    }
    // wave32 butterfly reduction
    ss += __shfl_xor(ss, 1);
    ss += __shfl_xor(ss, 2);
    ss += __shfl_xor(ss, 4);
    ss += __shfl_xor(ss, 8);
    ss += __shfl_xor(ss, 16);
    const float inv = 1.0f / (sqrtf(ss) + 1e-13f);

    _Float16* dst = outn + (size_t)row * EP_;
#pragma unroll
    for (int i = 0; i < 10; ++i)
        dst[lane + i * 32] = (_Float16)(v[i] * inv);
}

// ---------------------------------------------------------------------------
// Kernel 2: fused batched GEMM (WMMA f16 -> f32) + Gaussian kernel pooling +
// log/mask/dense reduction. One block per batch, 8 waves.
//   wave & 1  -> M tile (Q rows 0-15 / 16-31)
//   wave >> 1 -> N tile group; 8 N tiles of 16 per wave (covers D=512)
// Fragment layouts: 16-bit A 16x32 / B 32x16 use k = 16*(lane/16) + s in both; f32 C per CDNA5 ISA 7.12.2.
// B fragments are explicitly double-buffered so load(k+1) overlaps wmma(k).
// ---------------------------------------------------------------------------
__global__ __launch_bounds__(256)
void knrm_gemm_kernel(const _Float16* __restrict__ qn,
                      const _Float16* __restrict__ dn,
                      const float* __restrict__ qmask,
                      const float* __restrict__ dmask,
                      const float* __restrict__ mu,
                      const float* __restrict__ sigma,
                      const float* __restrict__ dense_w,
                      const float* __restrict__ dense_b,
                      float* __restrict__ out)
{
    __shared__ float s_pkq[Q_ * K_];   // per (q, kernel) sums over D
    __shared__ float s_red[256];

    const int b    = blockIdx.x;
    const int tid  = threadIdx.x;
    const int lane = tid & 31;
    const int wave = tid >> 5;
    const int lrow = lane & 15;        // N column within tile / M row select
    const int hs   = lane >> 4;        // lane-half: K-group select per ISA layout

    for (int i = tid; i < Q_ * K_; i += 256) s_pkq[i] = 0.0f;
    __syncthreads();

    // Gaussian kernel parameters (uniform per thread; L0-broadcast loads)
    float muv[K_], ninv[K_];
#pragma unroll
    for (int k = 0; k < K_; ++k) {
        muv[k] = mu[k];
        const float s = sigma[k];
        ninv[k] = -1.0f / (2.0f * s * s);
    }

    const int mtile = wave & 1;        // which 16 query rows
    const int ngrp  = wave >> 1;       // 0..3

    // A: row m = mtile*16 + lrow of q_n; two 8-half chunks per K-step.
    const _Float16* arow  = qn + ((size_t)b * Q_ + mtile * 16 + lrow) * EP_;
    const _Float16* dbase = dn + (size_t)b * D_ * EP_;

    // Preload all 10 A fragments (stay VGPR-resident across the N-tile loop).
    v16h afrag[EP_ / 32];
#pragma unroll
    for (int kk = 0; kk < EP_ / 32; ++kk) {
        const int k0 = kk * 32;
        const v8h a0 = *(const v8h*)(arow + k0 + hs * 16);
        const v8h a1 = *(const v8h*)(arow + k0 + hs * 16 + 8);
        afrag[kk] = __builtin_shufflevector(a0, a1,
            0, 1, 2, 3, 4, 5, 6, 7, 8, 9, 10, 11, 12, 13, 14, 15);
    }

    float qmv[8];
#pragma unroll
    for (int r = 0; r < 8; ++r)
        qmv[r] = qmask[b * Q_ + mtile * 16 + hs * 8 + r];

    float acc[8][K_];
#pragma unroll
    for (int r = 0; r < 8; ++r)
#pragma unroll
        for (int k = 0; k < K_; ++k) acc[r][k] = 0.0f;

    for (int j = 0; j < 8; ++j) {
        const int n0 = (ngrp + 4 * j) * 16;
        // B[k][n] = d_n[n][k]: lane holds column n = n0+lrow, K contiguous.
        const _Float16* brow = dbase + (size_t)(n0 + lrow) * EP_ + hs * 16;

        // Warm all cache levels (WGP scope) for the next tile's B row while
        // this tile computes: high-locality hint -> WGP-scope prefetch.
        if (j < 7) {
            const int n0n = (ngrp + 4 * (j + 1)) * 16;
            __builtin_prefetch(dbase + (size_t)(n0n + lrow) * EP_ + hs * 16, 0, 3);
        }

        v8f c = {};
        v16h bf = *(const v16h*)(brow);          // kk = 0 fragment
#pragma unroll
        for (int kk = 0; kk < EP_ / 32; ++kk) {
            v16h bf_next;
            if (kk < EP_ / 32 - 1)
                bf_next = *(const v16h*)(brow + (kk + 1) * 32);  // prefetch k+1
            else
                bf_next = bf;
            c = __builtin_amdgcn_wmma_f32_16x16x32_f16(
                    false, afrag[kk], false, bf, (short)0, c, false, false);
            bf = bf_next;
        }

        // C layout: lane<16 -> M = r, lane>=16 -> M = 8+r; N = lrow.
        const float dm = dmask[b * D_ + n0 + lrow];
#pragma unroll
        for (int r = 0; r < 8; ++r) {
            const float pm = qmv[r] * dm;
            const float cm = c[r] * pm;          // cosine * pair_mask
#pragma unroll
            for (int k = 0; k < K_; ++k) {
                const float diff = cm - muv[k];
                acc[r][k] += __expf(diff * diff * ninv[k]) * pm;
            }
        }
    }

    // Reduce over the 16 lanes of each half-wave (sums over N columns),
    // then accumulate into shared per-(q,k) sums.
#pragma unroll
    for (int r = 0; r < 8; ++r) {
#pragma unroll
        for (int k = 0; k < K_; ++k) {
            float v = acc[r][k];
            v += __shfl_xor(v, 1);
            v += __shfl_xor(v, 2);
            v += __shfl_xor(v, 4);
            v += __shfl_xor(v, 8);
            if (lrow == 0) {
                const int m = mtile * 16 + hs * 8 + r;
                atomicAdd(&s_pkq[m * K_ + k], v);
            }
        }
    }
    __syncthreads();

    // log / clip / q_mask / dense(W,b) -> scalar per batch
    float partial = 0.0f;
    for (int idx = tid; idx < Q_ * K_; idx += 256) {
        const int q = idx / K_;
        const int k = idx - q * K_;
        const float pkq = s_pkq[idx];
        partial += __logf(fmaxf(pkq, 1e-10f)) * 0.01f
                 * qmask[b * Q_ + q] * dense_w[k];
    }
    s_red[tid] = partial;
    __syncthreads();
    if (tid == 0) {
        float s = 0.0f;
        for (int i = 0; i < 256; ++i) s += s_red[i];
        out[b] = s + dense_b[0];
    }
}

// ---------------------------------------------------------------------------
// Launcher
// d_in: 0=query_tokens(int,[256,32]) 1=doc_tokens(int,[256,512])
//       2=emb_table(f32,[100000,300]) 3=dense_w(f32,[1,11]) 4=dense_b(f32,[1])
//       5=mu(f32,[11]) 6=sigma(f32,[11])
// d_out: f32[256]
// ---------------------------------------------------------------------------
extern "C" void kernel_launch(void* const* d_in, const int* in_sizes, int n_in,
                              void* d_out, int out_size, void* d_ws, size_t ws_size,
                              hipStream_t stream)
{
    const int*   qtok = (const int*)d_in[0];
    const int*   dtok = (const int*)d_in[1];
    const float* emb  = (const float*)d_in[2];
    const float* dw   = (const float*)d_in[3];
    const float* db   = (const float*)d_in[4];
    const float* mu   = (const float*)d_in[5];
    const float* sg   = (const float*)d_in[6];
    float*       out  = (float*)d_out;

    char* ws = (char*)d_ws;
    const size_t DN_BYTES = (size_t)B_ * D_ * EP_ * sizeof(_Float16); // 83.9 MB
    const size_t QN_BYTES = (size_t)B_ * Q_ * EP_ * sizeof(_Float16); //  5.2 MB
    const size_t DM_BYTES = (size_t)B_ * D_ * sizeof(float);
    _Float16* dn    = (_Float16*)(ws);
    _Float16* qn    = (_Float16*)(ws + DN_BYTES);
    float*    dmask = (float*)   (ws + DN_BYTES + QN_BYTES);
    float*    qmask = (float*)   (ws + DN_BYTES + QN_BYTES + DM_BYTES);

    // Gather + normalize: one wave per token row, 8 rows per block.
    knrm_normalize_kernel<<<(B_ * D_) / 8, 256, 0, stream>>>(dtok, emb, dn, dmask, B_ * D_);
    knrm_normalize_kernel<<<(B_ * Q_) / 8, 256, 0, stream>>>(qtok, emb, qn, qmask, B_ * Q_);

    // Fused WMMA GEMM + KNRM pooling + dense head. One block per batch.
    knrm_gemm_kernel<<<B_, 256, 0, stream>>>(qn, dn, qmask, dmask, mu, sg, dw, db, out);
}
